// SwinTransformerLayer_54236847014194
// MI455X (gfx1250) — compile-verified, matrix operand order checked by reading, not run on hardware
//
#include <hip/hip_runtime.h>
#include <hip/hip_bf16.h>
#include <math.h>

typedef __attribute__((ext_vector_type(16))) _Float16 v16h;
typedef __attribute__((ext_vector_type(8)))  _Float16 v8h;
typedef __attribute__((ext_vector_type(4)))  _Float16 v4h;
typedef __attribute__((ext_vector_type(8)))  float    v8f;
typedef __attribute__((ext_vector_type(4)))  float    v4f;

#define B_    32
#define H_    56
#define W_    56
#define C_    384
#define WS_   7
#define SS_   3
#define NH_   12
#define HD_   32
#define HID_  1536
#define NWIN  64           // 8x8 windows per image
#define NWTOT (B_ * NWIN)  // 2048
#define NWTOK 49
#define MTOT  (NWTOT * NWTOK)  // 100352 == B_*H_*W_
#define HW_   (H_ * W_)

// ---------------------------------------------------------------------------
// WMMA helpers (CDNA5 gfx1250, wave32)
// ---------------------------------------------------------------------------
__device__ __forceinline__ v8f wmma32(v16h a, v16h b, v8f c) {
  return __builtin_amdgcn_wmma_f32_16x16x32_f16(false, a, false, b, (short)0, c,
                                                false, false);
}

// Load a 16x32 (f16) A-fragment / 32x16 B-fragment from LDS.
// Lane lo holds row/col (idxBase+lo); per lane-group g, halves 0..7 are
// K=[kBase+8g, +8), halves 8..15 are K=[kBase+16+8g, +8).
// stride multiple of 8 halves; kBase multiple of 8 -> two aligned ds_load_b128.
__device__ __forceinline__ v16h frag_ld(const _Float16* __restrict__ tile,
                                        int idxBase, int kBase, int stride) {
  int l  = threadIdx.x & 31;
  int lo = l & 15;
  int g  = l >> 4;
  const _Float16* p = tile + (size_t)(idxBase + lo) * stride + kBase + 8 * g;
  union { v16h v; v8h h[2]; } u;
  u.h[0] = *(const v8h*)(p);
  u.h[1] = *(const v8h*)(p + 16);
  return u.v;
}

// ---------------------------------------------------------------------------
// One-time f32 -> f16 weight conversion (weights stay L2-resident)
// ---------------------------------------------------------------------------
__global__ __launch_bounds__(256) void cvt_f16(const float* __restrict__ s,
                                               _Float16* __restrict__ d, int n) {
  int i = (blockIdx.x * 256 + threadIdx.x) * 4;
  if (i < n) {
    v4f f = *(const v4f*)(s + i);
    v4h h;
    h[0] = (_Float16)f[0]; h[1] = (_Float16)f[1];
    h[2] = (_Float16)f[2]; h[3] = (_Float16)f[3];
    *(v4h*)(d + i) = h;
  }
}

// ---------------------------------------------------------------------------
// Window partition (roll(-3,-3) fused) + f32->f16 convert
// ---------------------------------------------------------------------------
__global__ __launch_bounds__(128) void prep_win(const float* __restrict__ x,
                                                _Float16* __restrict__ xw) {
  int row = blockIdx.x;              // windowed token row 0..MTOT-1
  int win = row / NWTOK, t = row % NWTOK;
  int b = win >> 6, wy = (win >> 3) & 7, wx = win & 7;
  int ty = t / WS_, tx = t % WS_;
  int hh = wy * WS_ + ty + SS_; if (hh >= H_) hh -= H_;
  int ww = wx * WS_ + tx + SS_; if (ww >= W_) ww -= W_;
  const float* src = x + ((size_t)b * HW_ + hh * W_ + ww) * C_;
  _Float16* dst = xw + (size_t)row * C_;
  for (int c = threadIdx.x; c < C_; c += 128) dst[c] = (_Float16)src[c];
}

// ---------------------------------------------------------------------------
// WMMA GEMM: C[M,N] = A[M,K](f16) * W[N,K]^T(f16) + bias, fused epilogues.
// Block: 256 threads (8 waves), tile 128(M) x 128(N), K chunk 64.
// Wave grid 4x2: each wave 32 rows x 64 cols -> acc[2][4]; 16 WMMA / chunk.
// EPI: 0=store f16 (qkv), 1=window-reverse scatter f32 (proj),
//      2=GELU->f16 (fc1), 3=residual add ->f32 (fc2 / output)
// ---------------------------------------------------------------------------
template <int EPI>
__global__ __launch_bounds__(256) void gemm_wmma(
    const _Float16* __restrict__ A, const _Float16* __restrict__ Wh,
    const float* __restrict__ bias, int K, int N,
    _Float16* __restrict__ outh, float* __restrict__ outf,
    const float* __restrict__ res) {
  __shared__ _Float16 sA[128 * 72];
  __shared__ _Float16 sB[128 * 72];

  const int tid = threadIdx.x;
  const int bm  = blockIdx.y * 128;
  const int bn  = blockIdx.x * 128;
  const int wid = tid >> 5;
  const int wm  = wid >> 1;   // 0..3 : 32 rows
  const int wn  = wid & 1;    // 0..1 : 64 cols

  v8f acc[2][4] = {};

  const int lr = tid >> 1;            // 0..127 : tile row for loads
  const int ls = (tid & 1) * 32;      // 0/32   : K segment

  for (int k0 = 0; k0 < K; k0 += 64) {
    // stage A and B tiles: 128x64 halves each, 4 x b128 per thread per tile
    {
      const v8h* sa = (const v8h*)(A + (size_t)(bm + lr) * K + k0 + ls);
      const v8h* sb = (const v8h*)(Wh + (size_t)(bn + lr) * K + k0 + ls);
      v8h* da = (v8h*)(sA + lr * 72 + ls);
      v8h* db = (v8h*)(sB + lr * 72 + ls);
#pragma unroll
      for (int i = 0; i < 4; ++i) { da[i] = sa[i]; db[i] = sb[i]; }
    }
    if (k0 + 64 < K) {  // prefetch next chunk (global_prefetch_b8)
      __builtin_prefetch(A + (size_t)(bm + lr) * K + k0 + 64 + ls, 0, 3);
      __builtin_prefetch(Wh + (size_t)(bn + lr) * K + k0 + 64 + ls, 0, 3);
    }
    __syncthreads();
#pragma unroll
    for (int ks = 0; ks < 64; ks += 32) {
      v16h a0 = frag_ld(sA, wm * 32 + 0, ks, 72);
      v16h a1 = frag_ld(sA, wm * 32 + 16, ks, 72);
#pragma unroll
      for (int tn = 0; tn < 4; ++tn) {
        v16h b = frag_ld(sB, wn * 64 + tn * 16, ks, 72);
        acc[0][tn] = wmma32(a0, b, acc[0][tn]);
        acc[1][tn] = wmma32(a1, b, acc[1][tn]);
      }
    }
    __syncthreads();
  }

  const int l = tid & 31, lo = l & 15, g = l >> 4;
#pragma unroll
  for (int tm = 0; tm < 2; ++tm)
#pragma unroll
    for (int tn = 0; tn < 4; ++tn) {
      int colb = bn + wn * 64 + tn * 16 + lo;
      float bv = bias[colb];
#pragma unroll
      for (int r = 0; r < 8; ++r) {
        int row = bm + wm * 32 + tm * 16 + r + 8 * g;
        float v = acc[tm][tn][r] + bv;
        if constexpr (EPI == 0) {
          outh[(size_t)row * N + colb] = (_Float16)v;
        } else if constexpr (EPI == 1) {
          // window reverse + roll(+3,+3), store f32 in spatial order
          int win = row / NWTOK, t = row % NWTOK;
          int b = win >> 6, wy = (win >> 3) & 7, wx = win & 7;
          int ty = t / WS_, tx = t % WS_;
          int hh = wy * WS_ + ty + SS_; if (hh >= H_) hh -= H_;
          int ww = wx * WS_ + tx + SS_; if (ww >= W_) ww -= W_;
          outf[((size_t)b * HW_ + hh * W_ + ww) * C_ + colb] = v;
        } else if constexpr (EPI == 2) {
          float gel = 0.5f * v * (1.0f + erff(v * 0.70710678118654752f));
          outh[(size_t)row * N + colb] = (_Float16)gel;
        } else {
          outf[(size_t)row * C_ + colb] = res[(size_t)row * C_ + colb] + v;
        }
      }
    }
}

// ---------------------------------------------------------------------------
// Windowed attention: one block per (head, window). 128 threads (4 waves).
// S = (Q*scale) K^T via WMMA, +rel-pos bias +shift mask (computed inline),
// row softmax, O = P V via WMMA.
// ---------------------------------------------------------------------------
__global__ __launch_bounds__(128) void attn_kernel(
    const _Float16* __restrict__ qkv, const float* __restrict__ rpb,
    _Float16* __restrict__ aout) {
  __shared__ _Float16 sQ[64 * 40];
  __shared__ _Float16 sK[64 * 40];
  __shared__ _Float16 sV[32 * 72];   // transposed: [hd][token]
  __shared__ _Float16 sP[64 * 72];
  __shared__ float    sS[64 * 65];

  const int h   = blockIdx.x;   // head
  const int win = blockIdx.y;   // window
  const int tid = threadIdx.x;
  const float scale = 0.17677669529663687f;  // HD^-0.5

  for (int i = tid; i < 64 * 32; i += 128) {
    int t = i >> 5, d = i & 31;
    float q = 0.f, k = 0.f, v = 0.f;
    if (t < NWTOK) {
      const _Float16* base = qkv + ((size_t)win * NWTOK + t) * (3 * C_) + h * HD_;
      q = (float)base[0 * C_ + d] * scale;
      k = (float)base[1 * C_ + d];
      v = (float)base[2 * C_ + d];
    }
    sQ[t * 40 + d] = (_Float16)q;
    sK[t * 40 + d] = (_Float16)k;
    sV[d * 72 + t] = (_Float16)v;
  }
  __syncthreads();

  const int wid = tid >> 5;
  const int l = tid & 31, lo = l & 15, g = l >> 4;

  // S = Q K^T : each wave owns a 16-row strip
  {
    v16h a = frag_ld(sQ, wid * 16, 0, 40);
#pragma unroll
    for (int nt = 0; nt < 4; ++nt) {
      v8f c = {};
      v16h b = frag_ld(sK, nt * 16, 0, 40);
      c = wmma32(a, b, c);
#pragma unroll
      for (int r = 0; r < 8; ++r)
        sS[(wid * 16 + r + 8 * g) * 65 + nt * 16 + lo] = c[r];
    }
  }
  __syncthreads();

  // softmax: one thread per valid row
  if (tid < NWTOK) {
    const int i = tid;
    const int ty = i / WS_, tx = i % WS_;
    const int wy = (win >> 3) & 7, wx = win & 7;
    const int gy = wy * WS_ + ty, gx = wx * WS_ + tx;
    const int ri = (gy < H_ - WS_ ? 0 : (gy < H_ - SS_ ? 1 : 2)) * 3 +
                   (gx < W_ - WS_ ? 0 : (gx < W_ - SS_ ? 1 : 2));
    float rowv[NWTOK];
    float mx = -1e30f;
    for (int j = 0; j < NWTOK; ++j) {
      int jy = j / WS_, jx = j % WS_;
      int hy = wy * WS_ + jy, hx = wx * WS_ + jx;
      int rj = (hy < H_ - WS_ ? 0 : (hy < H_ - SS_ ? 1 : 2)) * 3 +
               (hx < W_ - WS_ ? 0 : (hx < W_ - SS_ ? 1 : 2));
      int rel = (ty - jy + WS_ - 1) * (2 * WS_ - 1) + (tx - jx + WS_ - 1);
      float s = sS[i * 65 + j] + rpb[rel * NH_ + h] + (ri == rj ? 0.f : -100.f);
      rowv[j] = s;
      mx = fmaxf(mx, s);
    }
    float sum = 0.f;
    for (int j = 0; j < NWTOK; ++j) {
      float e = __expf(rowv[j] - mx);
      rowv[j] = e;
      sum += e;
    }
    float inv = 1.f / sum;
    for (int j = 0; j < NWTOK; ++j) sP[i * 72 + j] = (_Float16)(rowv[j] * inv);
    for (int j = NWTOK; j < 64; ++j) sP[i * 72 + j] = (_Float16)0.f;
  }
  // zero padded P rows 49..63 (their columns feed the K-dim of P*V)
  for (int i2 = NWTOK + tid; i2 < 64; i2 += 128)
    for (int j = 0; j < 64; ++j) sP[i2 * 72 + j] = (_Float16)0.f;
  __syncthreads();

  // O = P V
  v8f o0 = {}, o1 = {};
#pragma unroll
  for (int ks = 0; ks < 64; ks += 32) {
    v16h a  = frag_ld(sP, wid * 16, ks, 72);
    v16h b0 = frag_ld(sV, 0, ks, 72);
    v16h b1 = frag_ld(sV, 16, ks, 72);
    o0 = wmma32(a, b0, o0);
    o1 = wmma32(a, b1, o1);
  }
#pragma unroll
  for (int r = 0; r < 8; ++r) {
    int t = wid * 16 + r + 8 * g;
    if (t < NWTOK) {
      _Float16* dst = aout + ((size_t)win * NWTOK + t) * C_ + h * HD_;
      dst[lo]      = (_Float16)o0[r];
      dst[16 + lo] = (_Float16)o1[r];
    }
  }
}

// ---------------------------------------------------------------------------
// Fused: x1 = shortcut + LN1(xattn);  ln2h = f16(LN2(x1))
// ---------------------------------------------------------------------------
__device__ __forceinline__ float block_sum(float v, float* red) {
  red[threadIdx.x] = v;
  __syncthreads();
  for (int off = 64; off > 0; off >>= 1) {
    if ((int)threadIdx.x < off) red[threadIdx.x] += red[threadIdx.x + off];
    __syncthreads();
  }
  float r = red[0];
  __syncthreads();
  return r;
}

__global__ __launch_bounds__(128) void ln_kernel(
    const float* __restrict__ xattn, const float* __restrict__ x,
    const float* __restrict__ g1, const float* __restrict__ b1,
    const float* __restrict__ g2, const float* __restrict__ b2,
    float* __restrict__ x1, _Float16* __restrict__ ln2h) {
  __shared__ float red[128];
  const int row = blockIdx.x;
  const float* xa = xattn + (size_t)row * C_;
  const float* xi = x + (size_t)row * C_;
  const int c0 = threadIdx.x;

  float v[3];
  float s = 0.f;
#pragma unroll
  for (int i = 0; i < 3; ++i) { v[i] = xa[c0 + 128 * i]; s += v[i]; }
  float mu = block_sum(s, red) * (1.0f / C_);
  float s2 = 0.f;
#pragma unroll
  for (int i = 0; i < 3; ++i) { float d = v[i] - mu; s2 += d * d; }
  float rs = rsqrtf(block_sum(s2, red) * (1.0f / C_) + 1e-5f);

  float y[3];
  float sy = 0.f;
#pragma unroll
  for (int i = 0; i < 3; ++i) {
    int c = c0 + 128 * i;
    y[i] = xi[c] + (v[i] - mu) * rs * g1[c] + b1[c];
    x1[(size_t)row * C_ + c] = y[i];
    sy += y[i];
  }
  float mu2 = block_sum(sy, red) * (1.0f / C_);
  float s22 = 0.f;
#pragma unroll
  for (int i = 0; i < 3; ++i) { float d = y[i] - mu2; s22 += d * d; }
  float rs2 = rsqrtf(block_sum(s22, red) * (1.0f / C_) + 1e-5f);
#pragma unroll
  for (int i = 0; i < 3; ++i) {
    int c = c0 + 128 * i;
    ln2h[(size_t)row * C_ + c] = (_Float16)((y[i] - mu2) * rs2 * g2[c] + b2[c]);
  }
}

// ---------------------------------------------------------------------------
extern "C" void kernel_launch(void* const* d_in, const int* in_sizes, int n_in,
                              void* d_out, int out_size, void* d_ws,
                              size_t ws_size, hipStream_t stream) {
  (void)in_sizes; (void)n_in; (void)out_size; (void)ws_size;

  const float* x      = (const float*)d_in[0];
  const float* qkv_w  = (const float*)d_in[1];
  const float* qkv_b  = (const float*)d_in[2];
  const float* proj_w = (const float*)d_in[3];
  const float* proj_b = (const float*)d_in[4];
  const float* rpb    = (const float*)d_in[5];
  const float* g1     = (const float*)d_in[6];
  const float* b1     = (const float*)d_in[7];
  const float* g2     = (const float*)d_in[8];
  const float* b2     = (const float*)d_in[9];
  const float* fc1_w  = (const float*)d_in[10];
  const float* fc1_b  = (const float*)d_in[11];
  const float* fc2_w  = (const float*)d_in[12];
  const float* fc2_b  = (const float*)d_in[13];
  float* out = (float*)d_out;

  size_t off = 0;
  auto alloc = [&](size_t bytes) -> void* {
    void* p = (char*)d_ws + off;
    off += (bytes + 255) & ~(size_t)255;
    return p;
  };
  _Float16* xw     = (_Float16*)alloc((size_t)MTOT * C_ * 2);
  _Float16* qkvb   = (_Float16*)alloc((size_t)MTOT * 3 * C_ * 2);
  _Float16* aoutb  = (_Float16*)alloc((size_t)MTOT * C_ * 2);
  float*    xattn  = (float*)alloc((size_t)MTOT * C_ * 4);
  float*    x1     = (float*)alloc((size_t)MTOT * C_ * 4);
  _Float16* ln2h   = (_Float16*)alloc((size_t)MTOT * C_ * 2);
  _Float16* hbuf   = (_Float16*)alloc((size_t)MTOT * HID_ * 2);
  _Float16* qkv_wh = (_Float16*)alloc((size_t)3 * C_ * C_ * 2);
  _Float16* proj_wh= (_Float16*)alloc((size_t)C_ * C_ * 2);
  _Float16* fc1_wh = (_Float16*)alloc((size_t)HID_ * C_ * 2);
  _Float16* fc2_wh = (_Float16*)alloc((size_t)C_ * HID_ * 2);

  // 0) one-time weight conversions (L2-resident afterwards)
  cvt_f16<<<(3 * C_ * C_) / 1024, 256, 0, stream>>>(qkv_w, qkv_wh, 3 * C_ * C_);
  cvt_f16<<<(C_ * C_) / 1024, 256, 0, stream>>>(proj_w, proj_wh, C_ * C_);
  cvt_f16<<<(HID_ * C_) / 1024, 256, 0, stream>>>(fc1_w, fc1_wh, HID_ * C_);
  cvt_f16<<<(C_ * HID_) / 1024, 256, 0, stream>>>(fc2_w, fc2_wh, C_ * HID_);

  // 1) shifted window partition -> f16
  prep_win<<<MTOT, 128, 0, stream>>>(x, xw);
  // 2) qkv = xw @ qkv_w^T + b   [MTOT, 1152] f16
  gemm_wmma<0><<<dim3((3 * C_) / 128, MTOT / 128), 256, 0, stream>>>(
      xw, qkv_wh, qkv_b, C_, 3 * C_, qkvb, nullptr, nullptr);
  // 3) windowed attention  [MTOT, 384] f16
  attn_kernel<<<dim3(NH_, NWTOT), 128, 0, stream>>>(qkvb, rpb, aoutb);
  // 4) proj + window-reverse + unshift -> xattn f32 (spatial order)
  gemm_wmma<1><<<dim3(C_ / 128, MTOT / 128), 256, 0, stream>>>(
      aoutb, proj_wh, proj_b, C_, C_, nullptr, xattn, nullptr);
  // 5) x1 = x + LN1(xattn);  ln2h = LN2(x1) f16
  ln_kernel<<<MTOT, 128, 0, stream>>>(xattn, x, g1, b1, g2, b2, x1, ln2h);
  // 6) h = gelu(ln2h @ fc1^T + b)  f16
  gemm_wmma<2><<<dim3(HID_ / 128, MTOT / 128), 256, 0, stream>>>(
      ln2h, fc1_wh, fc1_b, C_, HID_, hbuf, nullptr, nullptr);
  // 7) out = x1 + h @ fc2^T + b   f32
  gemm_wmma<3><<<dim3(C_ / 128, MTOT / 128), 256, 0, stream>>>(
      hbuf, fc2_wh, fc2_b, HID_, C_, nullptr, out, x1);
}
